// gru_57449482551618
// MI455X (gfx1250) — matrix operand order repaired and checked by reading, not compiled, not run on hardware
//
#include <hip/hip_runtime.h>
#include <hip/hip_bf16.h>
#include <stdint.h>

// Problem sizes (match reference; max_seq == Sn)
#define Bn 64
#define Sn 512
#define En 1024
#define Dn 1024

typedef __attribute__((ext_vector_type(16))) __bf16 v16bf;
typedef __attribute__((ext_vector_type(8)))  float  v8f;

union Frag { v16bf v; uint4 u[2]; };

__device__ __forceinline__ unsigned short f32_to_bf16_bits(float f) {
    unsigned int u = __float_as_uint(f);
    unsigned int r = (u + 0x7FFFu + ((u >> 16) & 1u)) >> 16;  // round-to-nearest-even
    return (unsigned short)r;
}
__device__ __forceinline__ float sigmoidf_(float x) { return 1.0f / (1.0f + __expf(-x)); }

// ---------------------------------------------------------------------------
// f32 -> bf16 conversion (weights + x), grid-stride
// ---------------------------------------------------------------------------
__global__ void cvt_bf16_kernel(const float* __restrict__ in,
                                unsigned short* __restrict__ out, int n) {
    for (int i = blockIdx.x * blockDim.x + threadIdx.x; i < n;
         i += gridDim.x * blockDim.x)
        out[i] = f32_to_bf16_bits(in[i]);
}

__global__ void zero_h_kernel(float* __restrict__ hf, unsigned short* __restrict__ hbf) {
    int i = blockIdx.x * blockDim.x + threadIdx.x;
    if (i < Bn * Dn) { hf[i] = 0.0f; hbf[i] = 0; }
}

// ---------------------------------------------------------------------------
// Fragment loaders (wave32), contiguous k order used by this code for both A and B.
// A (16x32 bf16, M x K): lane = half*16 + m; element j -> k = half*16 + j
//   => one contiguous 16-half (32B) segment.
// B (32x16 bf16, K x N): lane = half*16 + n; element j -> k = half*16 + j
//   => one contiguous 16-half (32B) segment.
// For C = X @ W^T, B[k,n] = W[n,k]: lane n reads a contiguous row of W.
// ---------------------------------------------------------------------------
__device__ __forceinline__ Frag load_fragA(const unsigned short* __restrict__ rowPtr,
                                           int k0, int half) {
    Frag f;
    const unsigned short* p = rowPtr + k0 + half * 16;
    f.u[0] = *(const uint4*)(p);          // k = k0+half*16 .. +7
    f.u[1] = *(const uint4*)(p + 8);      // k = k0+half*16+8 .. +15
    return f;
}
__device__ __forceinline__ Frag load_fragB(const unsigned short* __restrict__ rowPtr,
                                           int k0, int half) {
    Frag f;
    const unsigned short* p = rowPtr + k0 + half * 16;
    f.u[0] = *(const uint4*)(p);
    f.u[1] = *(const uint4*)(p + 8);
    return f;
}

// ---------------------------------------------------------------------------
// Kernel 1: input projections  g{z,r,h}[s*B+b, d] = x[b,s,:] . W{z,r,h}[d,:] + bias
//
// Block = 256 threads (8 waves): 32-row (2 M-tiles) x 128-col (8 N-tiles) macro
// tile, all 3 gates. The 32x1024 bf16 A panel (64 KB) is staged into LDS once
// per block with per-lane global_load_async_to_lds_b128 (ASYNCcnt-tracked);
// the K-loop reads A from LDS (ds_load_b128) and B (weight rows) from L2,
// with next-iteration B fragments prefetched before the current WMMAs so the
// loadcnt waits never drain to zero.
// Grid: (2048/2 M-blocks) x (64/8 N-blocks) = 8192 blocks.
// ---------------------------------------------------------------------------
extern __shared__ unsigned short Ash[];   // 32 * 1024 bf16 = 64 KB

__global__ __launch_bounds__(256) void inproj_kernel(
    const unsigned short* __restrict__ xbf,
    const unsigned short* __restrict__ Wzb, const unsigned short* __restrict__ Wrb,
    const unsigned short* __restrict__ Whb,
    const float* __restrict__ pbz, const float* __restrict__ pbr,
    const float* __restrict__ pbh,
    float* __restrict__ gz, float* __restrict__ gr, float* __restrict__ gh) {
    const int nblk = blockIdx.x & 7;     // N macro-block (8 tn each)
    const int mblk = blockIdx.x >> 3;    // M macro-block (32 rows each)

    // ---- Stage A panel: 32 rows x 2048 B = 4096 x 16B chunks, 16 per thread ----
    {
        const int tId = threadIdx.x;
#pragma unroll
        for (int i = 0; i < 16; ++i) {
            const int c       = tId + i * 256;     // chunk id 0..4095
            const int r_local = c >> 7;            // 128 chunks per row
            const int within  = (c & 127) << 4;    // byte offset in row
            const int Mrow    = mblk * 32 + r_local;   // row = s*B + b
            const int s = Mrow >> 6, b = Mrow & 63;
            const char* gsrc =
                (const char*)(xbf + ((size_t)b * Sn + s) * En) + within;
            const unsigned int ldsOff =
                (unsigned int)(size_t)((const char*)Ash + (size_t)c * 16);
            asm volatile("global_load_async_to_lds_b128 %0, %1, off"
                         :: "v"(ldsOff), "v"(gsrc)
                         : "memory");
        }
        asm volatile("s_wait_asynccnt 0" ::: "memory");
        __syncthreads();
    }

    const int wv   = threadIdx.x >> 5;   // wave -> local tn
    const int lane = threadIdx.x & 31;
    const int ln   = lane & 15;
    const int half = lane >> 4;

    const int tn   = nblk * 8 + wv;
    const int dcol = tn * 16 + ln;
    const unsigned short* bzRow = Wzb + (size_t)dcol * En;
    const unsigned short* brRow = Wrb + (size_t)dcol * En;
    const unsigned short* bhRow = Whb + (size_t)dcol * En;

    // LDS A rows for this wave's two M-tiles
    const unsigned short* aRow0 = Ash + (size_t)(ln)      * En;
    const unsigned short* aRow1 = Ash + (size_t)(16 + ln) * En;

    v8f az0 = {}, ar0 = {}, ah0 = {};
    v8f az1 = {}, ar1 = {}, ah1 = {};

    // Software pipeline: B fragments prefetched one K-step ahead.
    Frag wz = load_fragB(bzRow, 0, half);
    Frag wr = load_fragB(brRow, 0, half);
    Frag wh = load_fragB(bhRow, 0, half);
    for (int k0 = 0; k0 < En; k0 += 32) {
        Frag a0 = load_fragA(aRow0, k0, half);   // ds_load_b128 x2
        Frag a1 = load_fragA(aRow1, k0, half);
        const int kn = (k0 + 32 < En) ? (k0 + 32) : k0;  // dummy reload on last
        Frag wzn = load_fragB(bzRow, kn, half);
        Frag wrn = load_fragB(brRow, kn, half);
        Frag whn = load_fragB(bhRow, kn, half);
        az0 = __builtin_amdgcn_wmma_f32_16x16x32_bf16(false, a0.v, false, wz.v,
                                                      (short)0, az0, false, false);
        az1 = __builtin_amdgcn_wmma_f32_16x16x32_bf16(false, a1.v, false, wz.v,
                                                      (short)0, az1, false, false);
        ar0 = __builtin_amdgcn_wmma_f32_16x16x32_bf16(false, a0.v, false, wr.v,
                                                      (short)0, ar0, false, false);
        ar1 = __builtin_amdgcn_wmma_f32_16x16x32_bf16(false, a1.v, false, wr.v,
                                                      (short)0, ar1, false, false);
        ah0 = __builtin_amdgcn_wmma_f32_16x16x32_bf16(false, a0.v, false, wh.v,
                                                      (short)0, ah0, false, false);
        ah1 = __builtin_amdgcn_wmma_f32_16x16x32_bf16(false, a1.v, false, wh.v,
                                                      (short)0, ah1, false, false);
        wz = wzn; wr = wrn; wh = whn;
    }

    // C layout: lane col = ln, VGPR r -> m = half*8 + r
    const float bzv = pbz[dcol], brv = pbr[dcol], bhv = pbh[dcol];
#pragma unroll
    for (int r = 0; r < 8; ++r) {
        const int m0 = mblk * 32 + half * 8 + r;        // tile 0 rows
        const int m1 = m0 + 16;                         // tile 1 rows
        const size_t i0 = (size_t)m0 * Dn + dcol;
        const size_t i1 = (size_t)m1 * Dn + dcol;
        gz[i0] = az0[r] + bzv;  gz[i1] = az1[r] + bzv;
        gr[i0] = ar0[r] + brv;  gr[i1] = ar1[r] + brv;
        gh[i0] = ah0[r] + bhv;  gh[i1] = ah1[r] + bhv;
    }
}

// ---------------------------------------------------------------------------
// Kernel 2a (per timestep): r,z gates.
//   r = sigmoid(gr_t + h @ Ur^T + cr); z = sigmoid(gz_t + h @ Uz^T + cz)
//   emits rh = bf16(r*h) for the Uh GEMM, and z (f32) for the update.
// M = B = 64 (4 tiles), N = D = 1024 (64 tiles) -> 256 wave tiles -> 32 blocks.
// U panels (4 MB bf16) are L2-resident across all 512 steps.
// ---------------------------------------------------------------------------
__global__ __launch_bounds__(256) void step_rz_kernel(
    const unsigned short* __restrict__ hbf, const float* __restrict__ hf,
    const unsigned short* __restrict__ Urb, const unsigned short* __restrict__ Uzb,
    const float* __restrict__ gr_t, const float* __restrict__ gz_t,
    const float* __restrict__ cr, const float* __restrict__ cz,
    unsigned short* __restrict__ rhbf, float* __restrict__ zbuf) {
    const int wave = threadIdx.x >> 5;
    const int lane = threadIdx.x & 31;
    const int tile = blockIdx.x * 8 + wave;
    const int tm   = tile & 3;
    const int tn   = tile >> 2;
    const int ln   = lane & 15;
    const int half = lane >> 4;

    const unsigned short* aRow = hbf + ((size_t)(tm * 16 + ln)) * Dn;
    const int dcol = tn * 16 + ln;
    const unsigned short* urRow = Urb + (size_t)dcol * Dn;
    const unsigned short* uzRow = Uzb + (size_t)dcol * Dn;

    v8f accr = {}, accz = {};
    Frag a  = load_fragA(aRow, 0, half);
    Frag ur = load_fragB(urRow, 0, half);
    Frag uz = load_fragB(uzRow, 0, half);
    for (int k0 = 0; k0 < Dn; k0 += 32) {
        const int kn = (k0 + 32 < Dn) ? (k0 + 32) : k0;
        Frag an  = load_fragA(aRow, kn, half);
        Frag urn = load_fragB(urRow, kn, half);
        Frag uzn = load_fragB(uzRow, kn, half);
        accr = __builtin_amdgcn_wmma_f32_16x16x32_bf16(false, a.v, false, ur.v,
                                                       (short)0, accr, false, false);
        accz = __builtin_amdgcn_wmma_f32_16x16x32_bf16(false, a.v, false, uz.v,
                                                       (short)0, accz, false, false);
        a = an; ur = urn; uz = uzn;
    }
    const float crv = cr[dcol], czv = cz[dcol];
#pragma unroll
    for (int r = 0; r < 8; ++r) {
        const int brow = tm * 16 + half * 8 + r;
        const size_t idx = (size_t)brow * Dn + dcol;
        const float rv = sigmoidf_(accr[r] + gr_t[idx] + crv);
        const float zv = sigmoidf_(accz[r] + gz_t[idx] + czv);
        rhbf[idx] = f32_to_bf16_bits(rv * hf[idx]);
        zbuf[idx] = zv;
    }
}

// ---------------------------------------------------------------------------
// Kernel 2b (per timestep): candidate + state update.
//   hn = tanh(gh_t + (r*h) @ Uh^T + ch);  h = (1-z)*h + z*hn; out[b,t,:] = h
// ---------------------------------------------------------------------------
__global__ __launch_bounds__(256) void step_h_kernel(
    const unsigned short* __restrict__ rhbf, const unsigned short* __restrict__ Uhb,
    const float* __restrict__ gh_t, const float* __restrict__ ch,
    const float* __restrict__ zbuf,
    float* __restrict__ hf, unsigned short* __restrict__ hbf,
    float* __restrict__ out, int t) {
    const int wave = threadIdx.x >> 5;
    const int lane = threadIdx.x & 31;
    const int tile = blockIdx.x * 8 + wave;
    const int tm   = tile & 3;
    const int tn   = tile >> 2;
    const int ln   = lane & 15;
    const int half = lane >> 4;

    const unsigned short* aRow = rhbf + ((size_t)(tm * 16 + ln)) * Dn;
    const int dcol = tn * 16 + ln;
    const unsigned short* uhRow = Uhb + (size_t)dcol * Dn;

    v8f acch = {};
    Frag a  = load_fragA(aRow, 0, half);
    Frag uh = load_fragB(uhRow, 0, half);
    for (int k0 = 0; k0 < Dn; k0 += 32) {
        const int kn = (k0 + 32 < Dn) ? (k0 + 32) : k0;
        Frag an  = load_fragA(aRow, kn, half);
        Frag uhn = load_fragB(uhRow, kn, half);
        acch = __builtin_amdgcn_wmma_f32_16x16x32_bf16(false, a.v, false, uh.v,
                                                       (short)0, acch, false, false);
        a = an; uh = uhn;
    }
    const float chv = ch[dcol];
#pragma unroll
    for (int r = 0; r < 8; ++r) {
        const int brow = tm * 16 + half * 8 + r;
        const size_t idx = (size_t)brow * Dn + dcol;
        const float hn = tanhf(acch[r] + gh_t[idx] + chv);
        const float zv = zbuf[idx];
        const float hv = (1.0f - zv) * hf[idx] + zv * hn;
        hf[idx]  = hv;
        hbf[idx] = f32_to_bf16_bits(hv);
        out[((size_t)brow * Sn + t) * Dn + dcol] = hv;   // out[b, t, d]
    }
}

// ---------------------------------------------------------------------------
// Host-side launch. Inputs (reference order):
//  0:x 1:Wz 2:bz 3:Wr 4:br 5:Wh 6:bh 7:Uz 8:cz 9:Ur 10:cr 11:Uh 12:ch 13:max_seq
// Workspace needs ~461 MB.
// ---------------------------------------------------------------------------
extern "C" void kernel_launch(void* const* d_in, const int* in_sizes, int n_in,
                              void* d_out, int out_size, void* d_ws, size_t ws_size,
                              hipStream_t stream) {
    const float* x  = (const float*)d_in[0];
    const float* Wz = (const float*)d_in[1];
    const float* bz = (const float*)d_in[2];
    const float* Wr = (const float*)d_in[3];
    const float* br = (const float*)d_in[4];
    const float* Wh = (const float*)d_in[5];
    const float* bh = (const float*)d_in[6];
    const float* Uz = (const float*)d_in[7];
    const float* cz = (const float*)d_in[8];
    const float* Ur = (const float*)d_in[9];
    const float* cr = (const float*)d_in[10];
    const float* Uh = (const float*)d_in[11];
    const float* ch = (const float*)d_in[12];
    float* out = (float*)d_out;

    // Workspace carve-up (256B aligned)
    char* w = (char*)d_ws;
    auto carve = [&](size_t bytes) -> void* {
        void* p = (void*)w;
        w += (bytes + 255) & ~(size_t)255;
        return p;
    };
    const size_t WE = (size_t)Dn * En;        // 1M elements per weight matrix
    unsigned short* Wzb = (unsigned short*)carve(WE * 2);
    unsigned short* Wrb = (unsigned short*)carve(WE * 2);
    unsigned short* Whb = (unsigned short*)carve(WE * 2);
    unsigned short* Uzb = (unsigned short*)carve(WE * 2);
    unsigned short* Urb = (unsigned short*)carve(WE * 2);
    unsigned short* Uhb = (unsigned short*)carve(WE * 2);
    const size_t XE = (size_t)Bn * Sn * En;   // 32M elements
    unsigned short* xbf = (unsigned short*)carve(XE * 2);
    const size_t GE = (size_t)Sn * Bn * Dn;   // 32M elements per gate buffer
    float* gz = (float*)carve(GE * 4);
    float* gr = (float*)carve(GE * 4);
    float* gh = (float*)carve(GE * 4);
    float*          hf   = (float*)carve((size_t)Bn * Dn * 4);
    unsigned short* hbf  = (unsigned short*)carve((size_t)Bn * Dn * 2);
    unsigned short* rhbf = (unsigned short*)carve((size_t)Bn * Dn * 2);
    float*          zbuf = (float*)carve((size_t)Bn * Dn * 4);

    // 1) Convert weights and x to bf16
    cvt_bf16_kernel<<<2048, 256, 0, stream>>>(Wz, Wzb, (int)WE);
    cvt_bf16_kernel<<<2048, 256, 0, stream>>>(Wr, Wrb, (int)WE);
    cvt_bf16_kernel<<<2048, 256, 0, stream>>>(Wh, Whb, (int)WE);
    cvt_bf16_kernel<<<2048, 256, 0, stream>>>(Uz, Uzb, (int)WE);
    cvt_bf16_kernel<<<2048, 256, 0, stream>>>(Ur, Urb, (int)WE);
    cvt_bf16_kernel<<<2048, 256, 0, stream>>>(Uh, Uhb, (int)WE);
    cvt_bf16_kernel<<<8192, 256, 0, stream>>>(x, xbf, (int)XE);
    zero_h_kernel<<<(Bn * Dn + 255) / 256, 256, 0, stream>>>(hf, hbf);

    // 2) Batched input projections: 8192 blocks, 64 KB dynamic LDS A-panel,
    //    async-to-LDS staging + ds_load fragments + pipelined 6 WMMA/K-iter.
    inproj_kernel<<<8192, 256, 64 * 1024, stream>>>(xbf, Wzb, Wrb, Whb,
                                                    bz, br, bh, gz, gr, gh);

    // 3) Sequential recurrence: 512 steps x (r,z GEMMs ; Uh GEMM + update)
    for (int t = 0; t < Sn; ++t) {
        const size_t off = (size_t)t * Bn * Dn;
        step_rz_kernel<<<32, 256, 0, stream>>>(hbf, hf, Urb, Uzb,
                                               gr + off, gz + off, cr, cz,
                                               rhbf, zbuf);
        step_h_kernel<<<32, 256, 0, stream>>>(rhbf, Uhb, gh + off, ch, zbuf,
                                              hf, hbf, out, t);
    }
}
